// ConditionalChannelProjection_77816217469185
// MI455X (gfx1250) — matrix operand order repaired and checked by reading, not compiled, not run on hardware
//
#include <hip/hip_runtime.h>
#include <math.h>

// ---------------------------------------------------------------------------
// Problem constants (from reference): B=32, C_in=C_out=256, H=W=64, E=1024
// ---------------------------------------------------------------------------
#define BATCH  32
#define C_IN   256
#define C_OUT  256
#define HW     4096     // 64*64
#define EDIM   1024
#define M_TILE 64       // pixels per block
#define LDA    264      // u16 row stride of A tiles in LDS (256 + 8 pad -> 4-bank step)
#define SRS    65       // f32 row stride of TDM stage (64 pix + 1 pad DWORD per row)
#define OST    68       // f32 row stride of epilogue buffer (64 + 4 pad)

typedef __bf16 bf16_t;
typedef __attribute__((ext_vector_type(16))) __bf16          v16bf;
typedef __attribute__((ext_vector_type(8)))  float           v8f;
typedef __attribute__((ext_vector_type(8)))  unsigned short  v8u16;
typedef __attribute__((ext_vector_type(16))) unsigned short  v16u16;
typedef __attribute__((ext_vector_type(4)))  unsigned int    v4u32;
typedef __attribute__((ext_vector_type(8)))  int             v8i32;
typedef __attribute__((ext_vector_type(4)))  int             v4i32;

// float -> bf16 (round to nearest even), manual so we never rely on __bf16 math
static __device__ __forceinline__ unsigned short f2bf(float f) {
  unsigned u = __float_as_uint(f);
  return (unsigned short)((u + 0x7FFFu + ((u >> 16) & 1u)) >> 16);
}
static __device__ __forceinline__ float bf2f(unsigned short h) {
  return __uint_as_float(((unsigned)h) << 16);
}

// Assemble a 16-element bf16 WMMA fragment from two 16B chunks (LDS or global).
static __device__ __forceinline__ v16bf mkfrag(const unsigned short* p0,
                                               const unsigned short* p1) {
  v8u16 a = *(const v8u16*)p0;
  v8u16 b = *(const v8u16*)p1;
  v16u16 c = __builtin_shufflevector(a, b, 0, 1, 2, 3, 4, 5, 6, 7,
                                     8, 9, 10, 11, 12, 13, 14, 15);
  return __builtin_bit_cast(v16bf, c);
}

static __device__ __forceinline__ float selu_f(float x) {
  const float lam = 1.0507009873554805f;
  const float al  = 1.6732632423543772f;
  return x > 0.0f ? lam * x : (lam * al) * (expf(x) - 1.0f);
}

// ---------------------------------------------------------------------------
// Kernel 1: adaLN modulation. mod = silu(emb) @ ada_w^T + ada_b  (32 x 512)
// rows 0..255 -> shift, rows 256..511 -> scale. Tiny (17 MFLOP), L2-resident.
// ---------------------------------------------------------------------------
__global__ __launch_bounds__(256) void adaln_mod_kernel(
    const float* __restrict__ emb, const float* __restrict__ ada_w,
    const float* __restrict__ ada_b, float* __restrict__ shiftg,
    float* __restrict__ scaleg) {
  __shared__ float semb[EDIM];
  int b = blockIdx.x, t = threadIdx.x;
  for (int i = t; i < EDIM; i += 256) {
    float e = emb[b * EDIM + i];
    semb[i] = e / (1.0f + expf(-e));   // SiLU
  }
  __syncthreads();
  float acc0 = ada_b[t];         // shift row t
  float acc1 = ada_b[t + 256];   // scale row t
  const float4* w0 = (const float4*)(ada_w + (size_t)t * EDIM);
  const float4* w1 = (const float4*)(ada_w + (size_t)(t + 256) * EDIM);
  const float4* s4 = (const float4*)semb;
  for (int j = 0; j < EDIM / 4; ++j) {
    float4 s = s4[j];
    float4 a = w0[j], c = w1[j];
    acc0 += a.x * s.x + a.y * s.y + a.z * s.z + a.w * s.w;
    acc1 += c.x * s.x + c.y * s.y + c.z * s.z + c.w * s.w;
  }
  shiftg[b * C_IN + t] = acc0;
  scaleg[b * C_IN + t] = acc1;
}

// ---------------------------------------------------------------------------
// Kernel 2: conv_w (C_out, C_in) f32 -> bf16 hi/lo, pre-swizzled into the
// B-fragment lane layout this GEMM uses (contiguous k) so the GEMM loop does contiguous b128 loads:
//   o = (((ntile*8 + kchunk)*32 + lane)*2 + half)*8 + e
//   N = ntile*16 + lane%16 ; K = kchunk*32 + (lane/16)*16 + half*8 + e
// ---------------------------------------------------------------------------
__global__ __launch_bounds__(256) void wprep_kernel(
    const float* __restrict__ conv_w, unsigned short* __restrict__ whi,
    unsigned short* __restrict__ wlo) {
  int o = blockIdx.x * 256 + threadIdx.x;  // 65536 elements
  int e  = o & 7;
  int h  = (o >> 3) & 1;
  int l  = (o >> 4) & 31;
  int kc = (o >> 9) & 7;
  int nt = o >> 12;
  int n = nt * 16 + (l & 15);
  int k = kc * 32 + (l >> 4) * 16 + h * 8 + e;
  float w = conv_w[n * C_IN + k];
  unsigned short hi = f2bf(w);
  unsigned short lo = f2bf(w - bf2f(hi));  // bf16 residual for split-GEMM
  whi[o] = hi;
  wlo[o] = lo;
}

// ---------------------------------------------------------------------------
// Kernel 3: fused RMSNorm + modulate + 1x1 conv (bf16x3 split WMMA) + SELU.
// One block = 64 consecutive pixels of one image, all 256 output channels.
// x slab arrives via the Tensor Data Mover (2D tile DMA, LDS row padding).
// 8 waves, each owns N-strip of 32 (2 n-tiles) x M=64 (4 m-tiles).
// ---------------------------------------------------------------------------
__global__ __launch_bounds__(256) void fused_kernel(
    const float* __restrict__ x, const float* __restrict__ conv_b,
    const float* __restrict__ shiftg, const float* __restrict__ scaleg,
    const unsigned short* __restrict__ whi,
    const unsigned short* __restrict__ wlo, float* __restrict__ y) {
  extern __shared__ __align__(16) char smem[];
  float* fac  = (float*)smem;        // 256 : 1 + scale[c]
  float* shf  = fac + 256;           // 256 : shift[c]
  float* psum = shf + 256;           // 256 : sumsq partials
  float* rmsv = psum + 256;          // 64  : rsqrt(mean(x^2)+eps) per pixel
  float* stage = rmsv + 64;          // 256 rows x 65 f32 : TDM tile (chan-major)
  unsigned short* Ahi = (unsigned short*)(stage + C_IN * SRS);   // 64 x 264
  unsigned short* Alo = Ahi + M_TILE * LDA;                      // 64 x 264
  float* outb = stage;               // epilogue alias (256 x 68 f32), dead-stage reuse

  const int t = threadIdx.x;
  const int blk = blockIdx.x;
  const int b = blk >> 6;                  // 64 blocks per image
  const int pixbase = (blk & 63) * M_TILE;

  // ---- TDM: DMA the 64-pixel x 256-channel x-slab into LDS ----------------
  // D# describes a 2D tile: tile_dim0=64 (contiguous pixels), tile_dim1=256
  // (channels, line stride 4096 elems), data_size=4B, LDS pad 1 DWORD per
  // 64-DWORD row -> stage row stride 65 (bank step 1, conflict-free).
  // TDM ignores EXEC, so gate on a wave-uniform scalar: only wave 0 issues.
  int wuni = __builtin_amdgcn_readfirstlane((int)threadIdx.x);  // == 32*wave
  if (wuni == 0) {
    unsigned long long ga =
        (unsigned long long)(const void*)(x + ((size_t)b * C_IN) * HW + pixbase);
    unsigned int lds_off = (unsigned int)(unsigned long long)(void*)stage;
    v4u32 g0;
    g0[0] = 1u;                                   // count=1 (valid descriptor)
    g0[1] = lds_off;                              // lds_addr (bytes)
    g0[2] = (unsigned int)ga;                     // global_addr[31:0]
    g0[3] = ((unsigned int)(ga >> 32) & 0x01FFFFFFu) | 0x80000000u;  // type=2
    v8i32 g1;
    g1[0] = (int)((2u << 16)        // data_size = 4B
                | (1u << 20)        // pad_enable
                | (5u << 22));      // pad_interval code 5 = 64 DWORDs
                                    // pad_amount code 0 = 1 DWORD
    g1[1] = (int)((unsigned)(HW & 0xFFFF) << 16);      // tensor_dim0[15:0]=4096
    g1[2] = (int)((unsigned)(C_IN & 0xFFFF) << 16);    // tensor_dim1[15:0]=256
    g1[3] = (int)((unsigned)M_TILE << 16);             // tile_dim0 = 64
    g1[4] = (int)(unsigned)C_IN;                       // tile_dim1 = 256
    g1[5] = (int)(unsigned)HW;                         // tensor_dim0_stride=4096
    g1[6] = 0;
    g1[7] = 0;
    v4i32 gz4 = {0, 0, 0, 0};                          // 2D: groups 2/3 unused
    v8i32 gz8 = {0, 0, 0, 0, 0, 0, 0, 0};              // extra group (6-arg form)
    __builtin_amdgcn_tensor_load_to_lds(g0, g1, gz4, gz4, gz8, 0);
    __builtin_amdgcn_s_wait_tensorcnt(0);
  }

  fac[t] = 1.0f + scaleg[b * C_IN + t];
  shf[t] = shiftg[b * C_IN + t];
  __syncthreads();   // TDM tile + fac/shf visible to all waves

  // ---- pass 1: sum of squares over channels (stage is channel-major) ------
  {
    int pix = t & 63;
    int cg = t >> 6;  // 4 channel groups
    float ss = 0.0f;
    for (int j = 0; j < 64; ++j) {
      int c = 4 * j + cg;
      float v = stage[c * SRS + pix];   // banks (c+pix)%64 -> conflict-free
      ss += v * v;
    }
    psum[t] = ss;  // t == cg*64 + pix
  }
  __syncthreads();
  if (t < 64) {
    float s = psum[t] + psum[t + 64] + psum[t + 128] + psum[t + 192];
    rmsv[t] = rsqrtf(s * (1.0f / 256.0f) + 1e-6f);
  }
  __syncthreads();

  // ---- pass 2: modulate + bf16 hi/lo split into WMMA-A layout -------------
  {
    float fc = fac[t], sh = shf[t];  // thread t owns channel t
    for (int j = 0; j < M_TILE; ++j) {
      float xm = fmaf(stage[t * SRS + j] * rmsv[j], fc, sh);
      unsigned short hi = f2bf(xm);
      Ahi[j * LDA + t] = hi;
      Alo[j * LDA + t] = f2bf(xm - bf2f(hi));
    }
  }
  __syncthreads();

  // ---- GEMM: D = A * W^T, split bf16 (3 WMMAs per tile, f32 accum) --------
  const int lane = t & 31;
  const int wv = t >> 5;        // wave id -> N base = 32*wv
  const int l15 = lane & 15;
  const int lh = lane >> 4;

  v8f acc[4][2];
#pragma unroll
  for (int mt = 0; mt < 4; ++mt)
#pragma unroll
    for (int nt = 0; nt < 2; ++nt)
      acc[mt][nt] = (v8f){0.f, 0.f, 0.f, 0.f, 0.f, 0.f, 0.f, 0.f};

  for (int kc = 0; kc < 8; ++kc) {
    int kk = kc * 32;
    // A fragments: lane l15 = row, half lh selects K sub-chunk [16h, 16h+16)
    v16bf Ah[4], Al[4];
#pragma unroll
    for (int mt = 0; mt < 4; ++mt) {
      int off = (mt * 16 + l15) * LDA + kk + 16 * lh;
      Ah[mt] = mkfrag(Ahi + off, Ahi + off + 8);
      Al[mt] = mkfrag(Alo + off, Alo + off + 8);
    }
#pragma unroll
    for (int nt = 0; nt < 2; ++nt) {
      int ntile = wv * 2 + nt;
      size_t base = ((size_t)((ntile * 8 + kc) * 32 + lane)) * 16;
      if (kc + 1 < 8) {  // hint next K-chunk of weights (L2-resident)
        __builtin_prefetch(whi + base + 32 * 16, 0, 1);
        __builtin_prefetch(wlo + base + 32 * 16, 0, 1);
      }
      v16bf Bh = mkfrag(whi + base, whi + base + 8);
      v16bf Bl = mkfrag(wlo + base, wlo + base + 8);
#pragma unroll
      for (int mt = 0; mt < 4; ++mt) {
        acc[mt][nt] = __builtin_amdgcn_wmma_f32_16x16x32_bf16(
            false, Ah[mt], false, Bh, (short)0, acc[mt][nt], false, false);
        acc[mt][nt] = __builtin_amdgcn_wmma_f32_16x16x32_bf16(
            false, Al[mt], false, Bh, (short)0, acc[mt][nt], false, false);
        acc[mt][nt] = __builtin_amdgcn_wmma_f32_16x16x32_bf16(
            false, Ah[mt], false, Bl, (short)0, acc[mt][nt], false, false);
      }
    }
  }
  __syncthreads();  // all waves done reading A before aliasing outb over it

  // ---- epilogue: bias + SELU, transpose via LDS, coalesced stores ---------
#pragma unroll
  for (int nt = 0; nt < 2; ++nt) {
    int n = wv * 32 + nt * 16 + l15;  // C/D layout: N = lane%16
    float bias = conv_b[n];
#pragma unroll
    for (int mt = 0; mt < 4; ++mt) {
      v8f a = acc[mt][nt];
      int pixl = mt * 16 + 8 * lh;    // C/D layout: M = 8*(lane/16) + r
      float4 v0, v1;
      v0.x = selu_f(a[0] + bias); v0.y = selu_f(a[1] + bias);
      v0.z = selu_f(a[2] + bias); v0.w = selu_f(a[3] + bias);
      v1.x = selu_f(a[4] + bias); v1.y = selu_f(a[5] + bias);
      v1.z = selu_f(a[6] + bias); v1.w = selu_f(a[7] + bias);
      *(float4*)&outb[n * OST + pixl]     = v0;
      *(float4*)&outb[n * OST + pixl + 4] = v1;
    }
  }
  __syncthreads();
  float* yb = y + (size_t)b * C_OUT * HW + pixbase;
  for (int j = 0; j < 16; ++j) {
    int n = j * 16 + (t >> 4);
    int p4 = (t & 15) * 4;
    float4 v = *(const float4*)&outb[n * OST + p4];
    *(float4*)&yb[(size_t)n * HW + p4] = v;
  }
}

// ---------------------------------------------------------------------------
extern "C" void kernel_launch(void* const* d_in, const int* in_sizes, int n_in,
                              void* d_out, int out_size, void* d_ws,
                              size_t ws_size, hipStream_t stream) {
  const float* x      = (const float*)d_in[0];
  const float* emb    = (const float*)d_in[1];
  const float* ada_w  = (const float*)d_in[2];
  const float* ada_b  = (const float*)d_in[3];
  const float* conv_w = (const float*)d_in[4];
  const float* conv_b = (const float*)d_in[5];
  float* y = (float*)d_out;

  // Workspace layout (rewritten every call; deterministic):
  //   [0,128K)    w_hi  (bf16, fragment-swizzled)
  //   [128K,256K) w_lo
  //   [256K,288K) shift (B x C f32)
  //   [288K,320K) scale (B x C f32)
  char* ws = (char*)d_ws;
  unsigned short* whi = (unsigned short*)ws;
  unsigned short* wlo = (unsigned short*)(ws + 131072);
  float* shiftg = (float*)(ws + 262144);
  float* scaleg = (float*)(ws + 294912);

  adaln_mod_kernel<<<BATCH, 256, 0, stream>>>(emb, ada_w, ada_b, shiftg, scaleg);
  wprep_kernel<<<(C_OUT * C_IN) / 256, 256, 0, stream>>>(conv_w, whi, wlo);

  // LDS: 832 f32 header + 256*65 f32 TDM stage + 2 * 64*264 u16 A tiles
  const size_t lds_bytes = (size_t)832 * 4 + (size_t)C_IN * SRS * 4 +
                           (size_t)2 * M_TILE * LDA * 2;   // = 137472 B
  (void)hipFuncSetAttribute(reinterpret_cast<const void*>(fused_kernel),
                            hipFuncAttributeMaxDynamicSharedMemorySize,
                            (int)lds_bytes);
  fused_kernel<<<(BATCH * HW) / M_TILE, 256, lds_bytes, stream>>>(
      x, conv_b, shiftg, scaleg, whi, wlo, y);
}
